// QKVFuse_18854906430150
// MI455X (gfx1250) — compile-verified, matrix operand order checked by reading, not run on hardware
//
#include <hip/hip_runtime.h>
#include <stdint.h>

typedef __attribute__((ext_vector_type(16))) __bf16 v16bf;
typedef __attribute__((ext_vector_type(8)))  float  v8f;
typedef __attribute__((ext_vector_type(4)))  unsigned int v4u;
typedef __attribute__((ext_vector_type(4)))  int  v4i;
typedef __attribute__((ext_vector_type(8)))  int  v8i;

#define NB    8
#define NC    192
#define HEAD  8
#define CPH   24
#define DD    3072     // feature dim per head-branch
#define LL    128      // sequence length per branch
#define KCH   96       // DD/32

__device__ __forceinline__ uint16_t f2bf(float x) {
    uint32_t u = __float_as_uint(x);
    uint32_t r = u + 0x7FFFu + ((u >> 16) & 1u);
    return (uint16_t)(r >> 16);
}
__device__ __forceinline__ float bf2f(uint16_t h) {
    return __uint_as_float(((uint32_t)h) << 16);
}
__device__ __forceinline__ v16bf load_frag(const uint16_t* p) {
    union { v16bf v; uint4 q[2]; } u;
    u.q[0] = *(const uint4*)(p);
    u.q[1] = *(const uint4*)(p + 16);
    return u.v;
}
// The TDM builtin is opaque to alias analysis; this forces the optimizer to
// treat LDS (escaped via ptrtoint) as modified so ds_loads are not folded.
__device__ __forceinline__ void lds_publish() { asm volatile("" ::: "memory"); }

// ---- Tensor Data Mover: 2D tile (bf16) global -> LDS, rows padded 64B+16B (stride 80B) ----
__device__ __forceinline__ void tdm2d(uint32_t lds_addr, const void* gp, v8i g1) {
    uint64_t ga = (uint64_t)(uintptr_t)gp;
    v4u g0 = { 1u,                                   // count=1, user mode
               lds_addr,                             // LDS byte offset
               (uint32_t)ga,                         // global_addr[31:0]
               (uint32_t)((ga >> 32) & 0x01FFFFFFu) | (2u << 30) }; // addr[56:32] | type=2
    v4i z4 = { 0, 0, 0, 0 };
#if defined(__clang_major__) && __clang_major__ >= 23
    v8i z8 = { 0, 0, 0, 0, 0, 0, 0, 0 };
    __builtin_amdgcn_tensor_load_to_lds(g0, g1, z4, z4, z8, 0);
#else
    __builtin_amdgcn_tensor_load_to_lds(g0, g1, z4, z4, 0);
#endif
    asm volatile("" ::: "memory");
}
__device__ __forceinline__ uint32_t lds_off(const void* p) {
    return (uint32_t)(uintptr_t)p;                   // low 32b of generic LDS ptr = LDS offset
}

// -------- 1) convert inputs to bf16 in to_h / to_w layouts --------
__global__ __launch_bounds__(256) void prepass_kernel(
    const float* __restrict__ out1, const float* __restrict__ out2,
    uint16_t* __restrict__ Ah1, uint16_t* __restrict__ Ah2,
    uint16_t* __restrict__ Aw1, uint16_t* __restrict__ Aw2) {
    int idx = blockIdx.x * 256 + threadIdx.x;
    int ww = idx & 127;
    int hh = (idx >> 7) & 127;
    int ch = (idx >> 14) % NC;
    int b  = idx / (NC * 128 * 128);
    int n  = ch / CPH, cp = ch - n * CPH;
    int bn = b * HEAD + n;
    size_t ih = (size_t)(bn * LL + hh) * DD + (size_t)ww * CPH + cp;
    size_t iw = (size_t)(bn * LL + ww) * DD + (size_t)hh * CPH + cp;
    float v1 = out1[idx], v2 = out2[idx];
    Ah1[ih] = f2bf(v1); Ah2[ih] = f2bf(v2);
    Aw1[iw] = f2bf(v1); Aw2[iw] = f2bf(v2);
}

// -------- 2) reciprocal row norms --------
__global__ __launch_bounds__(256) void rnorm_kernel(
    const uint16_t* __restrict__ t0, const uint16_t* __restrict__ t1,
    const uint16_t* __restrict__ t2, const uint16_t* __restrict__ t3,
    float* __restrict__ rn) {
    __shared__ float red[256];
    const uint16_t* t = (blockIdx.y == 0) ? t0 : (blockIdx.y == 1) ? t1
                      : (blockIdx.y == 2) ? t2 : t3;
    size_t base = (size_t)blockIdx.x * DD;
    float s = 0.f;
    for (int k = threadIdx.x; k < DD; k += 256) { float f = bf2f(t[base + k]); s += f * f; }
    red[threadIdx.x] = s; __syncthreads();
    for (int off = 128; off > 0; off >>= 1) {
        if ((int)threadIdx.x < off) red[threadIdx.x] += red[threadIdx.x + off];
        __syncthreads();
    }
    if (threadIdx.x == 0)
        rn[blockIdx.y * 8192 + blockIdx.x] = 1.0f / fmaxf(sqrtf(red[0]), 1e-12f);
}

// -------- 3) attention: block = (b*head, 32-row slab); TDM-fed WMMA --------
#define OFF_QS0 0u
#define OFF_QS1 2560u
#define OFF_KS0 5120u
#define OFF_KS1 15360u
#define OFF_UN  25600u   // Ss (phase2, 16896B f32) union Vt (phase3, 17408B bf16)
#define OFF_PB  43008u
#define OFF_RQ  51712u
#define OFF_RK  51840u
#define SMEM_SZ 52352u

__global__ __launch_bounds__(256) void attn_kernel(
    const uint16_t* __restrict__ Qg, const uint16_t* __restrict__ KVg,
    const float* __restrict__ rqg, const float* __restrict__ rkg,
    uint16_t* __restrict__ Og) {
    __shared__ __align__(16) uint8_t smem[SMEM_SZ];
    float*    Ss  = (float*)(smem + OFF_UN);
    uint16_t* Vt  = (uint16_t*)(smem + OFF_UN);
    uint16_t* Pb  = (uint16_t*)(smem + OFF_PB);
    float*    rqs = (float*)(smem + OFF_RQ);
    float*    rks = (float*)(smem + OFF_RK);
    uint16_t* Qc  = (uint16_t*)(smem + OFF_KS0);   // phase-3 Q chunk (Ks dead then)

    const int tid  = threadIdx.x;
    const int wave = tid >> 5, lane = tid & 31;
    const int ln = lane & 15, hi = lane >> 4;
    const int bn = blockIdx.x >> 2;
    const int r_base = (blockIdx.x & 3) * 32;
    const int rt = wave & 1;
    const int jq = wave >> 1;

    if (tid < 32)  rqs[tid] = rqg[bn * 128 + r_base + tid];
    if (tid < 128) rks[tid] = rkg[bn * 128 + tid];

    const uint16_t* Kgp = KVg + (size_t)bn * 128 * DD;
    const uint16_t* Qgp = Qg  + (size_t)(bn * 128 + r_base) * DD;

    // TDM group1: data_size=2B, pad_enable, pad every 64B by 16B (LDS row stride 80B),
    // huge tensor dims (no clipping), tile_dim0=32 elems, row stride DD elems.
    const int PADW0 = (int)((1u << 16) | (1u << 20) | (3u << 22) | (3u << 25));
    v8i g1k = { PADW0, 0, 0x0010, 0x00200010, 128, (int)DD, 0, 0 };
    v8i g1q = { PADW0, 0, 0x0010, 0x00200010, 32,  (int)DD, 0, 0 };

    v8f zero = {0.f, 0.f, 0.f, 0.f, 0.f, 0.f, 0.f, 0.f};
    v8f acc0 = zero, acc1 = zero;

    // ---- phase 1: S = Q.K^T, double-buffered TDM pipeline ----
    if (wave == 0) {
        tdm2d(OFF_KS0 + lds_off(smem), Kgp, g1k);
        tdm2d(OFF_QS0 + lds_off(smem), Qgp, g1q);
    }
    for (int kc = 0; kc < KCH; ++kc) {
        int cb = kc & 1;
        if (wave == 0) {
            int kn = (kc + 1 < KCH) ? kc + 1 : 0;
            tdm2d((cb ? OFF_KS0 : OFF_KS1) + lds_off(smem), Kgp + (size_t)kn * 32, g1k);
            tdm2d((cb ? OFF_QS0 : OFF_QS1) + lds_off(smem), Qgp + (size_t)kn * 32, g1q);
            __builtin_amdgcn_s_wait_tensorcnt(2);   // oldest chunk (this iter's) done
        }
        __syncthreads();
        lds_publish();                              // TDM wrote LDS: keep ds_loads real
        const uint16_t* Qsc = (const uint16_t*)(smem + (cb ? OFF_QS1 : OFF_QS0));
        const uint16_t* Ksc = (const uint16_t*)(smem + (cb ? OFF_KS1 : OFF_KS0));
        v16bf a  = load_frag(&Qsc[(rt * 16 + ln) * 40 + hi * 8]);
        v16bf b0 = load_frag(&Ksc[(jq * 32 + ln) * 40 + hi * 8]);
        acc0 = __builtin_amdgcn_wmma_f32_16x16x32_bf16(false, a, false, b0, (short)0, acc0, false, false);
        v16bf b1 = load_frag(&Ksc[(jq * 32 + 16 + ln) * 40 + hi * 8]);
        acc1 = __builtin_amdgcn_wmma_f32_16x16x32_bf16(false, a, false, b1, (short)0, acc1, false, false);
        __syncthreads();
    }
    if (wave == 0) __builtin_amdgcn_s_wait_tensorcnt(0);  // drain wrapped prefetch
    lds_publish();

    // ---- phase 2: scale by reciprocal norms, softmax, P -> bf16 ----
#pragma unroll
    for (int g = 0; g < 8; ++g) {
        int il = rt * 16 + g + hi * 8;
        int j0 = jq * 32 + ln;
        Ss[il * 132 + j0]      = acc0[g] * rqs[il] * rks[j0];
        Ss[il * 132 + j0 + 16] = acc1[g] * rqs[il] * rks[j0 + 16];
    }
    __syncthreads();
    if (tid < 32) {
        float m = -3.4e38f;
        for (int j = 0; j < 128; ++j) m = fmaxf(m, Ss[tid * 132 + j]);
        float s = 0.f;
        for (int j = 0; j < 128; ++j) { float e = __expf(Ss[tid * 132 + j] - m); Ss[tid * 132 + j] = e; s += e; }
        float inv = 1.0f / s;
        for (int j = 0; j < 128; ++j) Pb[tid * 136 + j] = f2bf(Ss[tid * 132 + j] * inv);
    }

    // ---- phase 3: O = P.V + q1, 48 chunks of 64 d, ping-pong register buffers ----
    uint16_t* Outp = Og + (size_t)bn * 128 * DD;
    const uint4* Kg4 = (const uint4*)Kgp;
    const uint4* Qg4 = (const uint4*)Qgp;

    auto issue = [&](int d, uint4* rv, uint4& rq) {
#pragma unroll
        for (int s = 0; s < 4; ++s) {               // V: 128 rows x 8 segs, 4/thread
            int seg = tid * 4 + s;
            int row = seg >> 3, q = seg & 7;
            rv[s] = Kg4[(size_t)row * 384 + d * 8 + q];
        }
        int row = tid >> 3, q = tid & 7;            // Q: 32 rows x 8 segs, 1/thread
        rq = Qg4[(size_t)row * 384 + d * 8 + q];
    };
    auto stage = [&](const uint4* rv, const uint4& rq) {
#pragma unroll
        for (int s = 0; s < 4; ++s) {
            int seg = tid * 4 + s;
            int row = seg >> 3, q = seg & 7;
            const uint16_t* pv = (const uint16_t*)&rv[s];
#pragma unroll
            for (int e = 0; e < 8; ++e) Vt[(q * 8 + e) * 136 + row] = pv[e];
        }
        int row = tid >> 3, q = tid & 7;
        *(uint4*)(&Qc[row * 72 + q * 8]) = rq;
    };
    auto compute = [&](int dcc) {
        v8f oacc = zero;
#pragma unroll
        for (int kk = 0; kk < 4; ++kk) {
            v16bf a = load_frag(&Pb[(rt * 16 + ln) * 136 + kk * 32 + hi * 8]);
            v16bf b = load_frag(&Vt[(jq * 16 + ln) * 136 + kk * 32 + hi * 8]);
            oacc = __builtin_amdgcn_wmma_f32_16x16x32_bf16(false, a, false, b, (short)0, oacc, false, false);
        }
#pragma unroll
        for (int g = 0; g < 8; ++g) {
            int il = rt * 16 + g + hi * 8;
            int dl = jq * 16 + ln;
            float q1 = bf2f(Qc[il * 72 + dl]) * rqs[il];
            Outp[(size_t)(r_base + il) * DD + dcc * 64 + dl] = f2bf(oacc[g] + q1);
        }
    };

    uint4 rvA[4], rvB[4], rqA, rqB;
    issue(0, rvA, rqA);
    for (int dcc = 0; dcc < 48; dcc += 2) {
        __syncthreads();
        stage(rvA, rqA);
        issue(dcc + 1, rvB, rqB);                   // overlaps compute(dcc)
        __syncthreads();
        compute(dcc);
        __syncthreads();
        stage(rvB, rqB);
        issue(dcc + 2 < 48 ? dcc + 2 : 0, rvA, rqA);
        __syncthreads();
        compute(dcc + 1);
    }
}

// -------- 4) fused projection: out = W.(o3+o4) + 2*bias + out1 + out2 --------
__global__ __launch_bounds__(256) void proj_kernel(
    const uint16_t* __restrict__ O3, const uint16_t* __restrict__ O4,
    const float* __restrict__ Wm, const float* __restrict__ bias,
    const float* __restrict__ out1, const float* __restrict__ out2,
    float* __restrict__ outp) {
    __shared__ __align__(16) uint16_t Ws[192 * 200];
    __shared__ __align__(16) uint16_t Xt[128 * 200];
    const int tid = threadIdx.x;
    const int wave = tid >> 5, lane = tid & 31;
    const int ln = lane & 15, hi = lane >> 4;
    const int b = blockIdx.x >> 7, hh = blockIdx.x & 127;

    for (int e = tid; e < 192 * 192; e += 256) {
        int o = e / 192, c = e - o * 192;
        Ws[o * 200 + c] = f2bf(Wm[e]);
    }
    for (int e = tid; e < 192 * 128; e += 256) {
        int c = e >> 7, ww = e & 127;
        int n = c / CPH, cp = c - n * CPH;
        int bn = b * HEAD + n;
        float v = bf2f(O3[(size_t)(bn * 128 + hh) * DD + (size_t)ww * CPH + cp])
                + bf2f(O4[(size_t)(bn * 128 + ww) * DD + (size_t)hh * CPH + cp]);
        Xt[ww * 200 + c] = f2bf(v);
    }
    __syncthreads();

    v8f zero = {0.f, 0.f, 0.f, 0.f, 0.f, 0.f, 0.f, 0.f};
    for (int job = wave; job < 96; job += 8) {
        int mt = job >> 3, nt = job & 7;
        v8f acc = zero;
#pragma unroll
        for (int kk = 0; kk < 6; ++kk) {
            v16bf a = load_frag(&Ws[(mt * 16 + ln) * 200 + kk * 32 + hi * 8]);
            v16bf x = load_frag(&Xt[(nt * 16 + ln) * 200 + kk * 32 + hi * 8]);
            acc = __builtin_amdgcn_wmma_f32_16x16x32_bf16(false, a, false, x, (short)0, acc, false, false);
        }
#pragma unroll
        for (int g = 0; g < 8; ++g) {
            int o = mt * 16 + g + hi * 8;
            int ww = nt * 16 + ln;
            size_t gi = ((size_t)(b * NC + o) * 128 + hh) * 128 + ww;
            outp[gi] = acc[g] + 2.0f * bias[o] + out1[gi] + out2[gi];
        }
    }
}

extern "C" void kernel_launch(void* const* d_in, const int* in_sizes, int n_in,
                              void* d_out, int out_size, void* d_ws, size_t ws_size,
                              hipStream_t stream) {
    const float* out1 = (const float*)d_in[0];
    const float* out2 = (const float*)d_in[1];
    const float* Wm   = (const float*)d_in[2];
    const float* bias = (const float*)d_in[3];
    float* outp = (float*)d_out;

    const size_t TEN = (size_t)NB * HEAD * LL * DD;
    uint16_t* Ah1 = (uint16_t*)d_ws;
    uint16_t* Ah2 = Ah1 + TEN;
    uint16_t* Aw1 = Ah2 + TEN;
    uint16_t* Aw2 = Aw1 + TEN;
    uint16_t* O3  = Aw2 + TEN;
    uint16_t* O4  = O3 + TEN;
    float*    rn  = (float*)(O4 + TEN);

    prepass_kernel<<<98304, 256, 0, stream>>>(out1, out2, Ah1, Ah2, Aw1, Aw2);
    rnorm_kernel<<<dim3(8192, 4), 256, 0, stream>>>(Ah1, Ah2, Aw1, Aw2, rn);
    attn_kernel<<<256, 256, 0, stream>>>(Ah2, Ah1, rn + 8192, rn + 0, O3);       // branch 1
    attn_kernel<<<256, 256, 0, stream>>>(Aw1, Aw2, rn + 16384, rn + 24576, O4);  // branch 2
    proj_kernel<<<1024, 256, 0, stream>>>(O3, O4, Wm, bias, out1, out2, outp);
}
